// SelfAttention_343597383821
// MI455X (gfx1250) — matrix operand order repaired and checked by reading, not compiled, not run on hardware
//
#include <hip/hip_runtime.h>
#include <hip/hip_bf16.h>
#include <cmath>
#include <cstdint>

// ---------------- problem constants ----------------
constexpr int BATCH = 4;
constexpr int SEQ   = 2048;
constexpr int DIN   = 1024;
constexpr int DH    = 1024;
constexpr long long BS = (long long)BATCH * SEQ;   // 8192 rows

// ---------------- WMMA / TDM vector types ----------------
typedef __attribute__((ext_vector_type(16))) __bf16 v16bf;
typedef __attribute__((ext_vector_type(8)))  float  v8f;
typedef unsigned int v4u  __attribute__((ext_vector_type(4)));
typedef int          v4i_t __attribute__((ext_vector_type(4)));
typedef int          v8i_t __attribute__((ext_vector_type(8)));

#if defined(__AMDGCN__) && __has_builtin(__builtin_amdgcn_tensor_load_to_lds) && \
    __has_builtin(__builtin_amdgcn_s_wait_tensorcnt)
#define USE_TDM 1
#else
#define USE_TDM 0
#endif

// ---------------- helpers ----------------
__device__ __forceinline__ unsigned short f32_to_bf16(float f) {
    unsigned int u = __float_as_uint(f);
    u = u + 0x7FFFu + ((u >> 16) & 1u);   // round-to-nearest-even
    return (unsigned short)(u >> 16);
}

// LDS tile strides (padded to dodge bank conflicts)
constexpr int BM = 128, BN = 128, BK = 32;
constexpr int LDA_S = BK + 4;   // 36 halves per A row  (64B row + 8B pad)
constexpr int LDB_S = BN + 8;   // 136 halves per B row (256B row + 16B pad)

#if USE_TDM
// Issue one 2D tile DMA: global (row-major, stride0 elements between rows)
// -> LDS row-major with (pad_amount+1) DWORDs of padding after every
// 2^(pad_interval+1) DWORDs.  Descriptor per CDNA5 ISA.
__device__ __forceinline__ void tdm_load_2d(uint32_t lds_byte_addr, const void* gptr,
                                            uint32_t tile_d0, uint32_t tile_d1,
                                            uint32_t tensor_d0, uint32_t tensor_d1,
                                            uint64_t stride0,
                                            uint32_t pad_interval_code,
                                            uint32_t pad_amount_code) {
    const uint64_t ga = (uint64_t)(uintptr_t)gptr;
    v4u g0;
    g0.x = 1u;                                                  // count=1, user mode
    g0.y = lds_byte_addr;                                       // lds_addr [63:32]
    g0.z = (uint32_t)(ga & 0xFFFFFFFFu);                        // global_addr [95:64]
    g0.w = (uint32_t)((ga >> 32) & 0x01FFFFFFu) | 0x80000000u;  // addr[56:32] | type=2

    v8i_t g1;
    const uint32_t flags = (1u << 16)                 // data_size = 2 bytes
                         | (1u << 20)                 // pad_enable
                         | (pad_interval_code << 22)
                         | (pad_amount_code << 25);
    g1[0] = (int)flags;                                          // mask=0 | flags
    g1[1] = (int)((tensor_d0 & 0xFFFFu) << 16);                  // tensor_dim0[15:0]
    g1[2] = (int)((tensor_d0 >> 16) | ((tensor_d1 & 0xFFFFu) << 16));
    g1[3] = (int)((tensor_d1 >> 16) | (tile_d0 << 16));          // tile_dim0
    g1[4] = (int)(tile_d1 & 0xFFFFu);                            // tile_dim1 | tile_dim2=0
    g1[5] = (int)(uint32_t)(stride0 & 0xFFFFFFFFu);              // dim0_stride lo
    g1[6] = (int)(uint32_t)((stride0 >> 32) & 0xFFFFu);          // dim0_stride hi | dim1_stride=0
    g1[7] = 0;

    const v4i_t z4 = {0, 0, 0, 0};
#if __clang_major__ >= 23
    const v8i_t z8 = {0, 0, 0, 0, 0, 0, 0, 0};
    __builtin_amdgcn_tensor_load_to_lds(g0, g1, z4, z4, z8, 0);
#else
    __builtin_amdgcn_tensor_load_to_lds(g0, g1, z4, z4, 0);
#endif
}
#endif  // USE_TDM

// A fragment: 16x32 bf16.  Lane L%16 = row M, K-halves per ISA table:
//   VGPR j<4 -> K = 2j (+8 for lanes 16..31); j>=4 -> K = 2j+8 (+8).
__device__ __forceinline__ v16bf ld_frag_a(const unsigned short* As, int rowBase, int lane) {
    union { v16bf v; uint32_t w[8]; } f;
    const unsigned short* p = As + (rowBase + (lane & 15)) * LDA_S;
    const int kh = (lane >> 4) << 3;   // 0 or 8
#pragma unroll
    for (int j = 0; j < 8; ++j) {
        const int kb = ((j < 4) ? (2 * j) : (2 * j + 8)) + kh;
        f.w[j] = *(const uint32_t*)(p + kb);
    }
    return f.v;
}

// B fragment: 32x16 bf16 (K x N).  Lane L%16 = col N; K-halves same as A fragment.
__device__ __forceinline__ v16bf ld_frag_b(const unsigned short* Bs, int colBase, int lane) {
    union { v16bf v; uint32_t w[8]; } f;
    const unsigned short* p = Bs + colBase + (lane & 15);
    const int kh = (lane >> 4) << 3;   // 0 or 8
#pragma unroll
    for (int j = 0; j < 8; ++j) {
        const int kb = ((j < 4) ? (2 * j) : (2 * j + 8)) + kh;
        f.w[j] = (uint32_t)p[kb * LDB_S] | ((uint32_t)p[(kb + 1) * LDB_S] << 16);
    }
    return f.v;
}

// ---------------- prep kernels ----------------
__global__ void f32_to_bf16_kernel(const float* __restrict__ in,
                                   unsigned short* __restrict__ out, long long n) {
    long long i = (long long)blockIdx.x * blockDim.x + threadIdx.x;
    if (i < n) out[i] = f32_to_bf16(in[i]);
}

// W is [DH][DIN] row-major fp32; produce Wt[d][h] = W[h][d] in bf16.
__global__ void transpose_w_kernel(const float* __restrict__ W,
                                   unsigned short* __restrict__ Wt) {
    __shared__ float tile[32][33];
    const int bx = blockIdx.x * 32;   // d block
    const int by = blockIdx.y * 32;   // h block
    const int tx = threadIdx.x, ty = threadIdx.y;  // (32, 8)
#pragma unroll
    for (int i = 0; i < 32; i += 8)
        tile[ty + i][tx] = W[(long long)(by + ty + i) * DIN + bx + tx];
    __syncthreads();
#pragma unroll
    for (int i = 0; i < 32; i += 8)
        Wt[(long long)(bx + ty + i) * DH + by + tx] = f32_to_bf16(tile[tx][ty + i]);
}

// ---------------- batched WMMA GEMM ----------------
// C[b] = scale * (A[b] @ op(B[b]))
//   A: bf16 row-major M x K (lda).  Staged to LDS by the Tensor Data Mover.
//   B: BT==false -> bf16 K x N (ldb), staged by TDM; BT==true -> bf16 N x K (ldb),
//      staged manually with an in-flight transpose (TDM cannot transpose).
//   C: OUT_BF16 ? bf16 : f32, row-major M x N (ldc)
// Block tile 128x128, 8 waves (4 in M x 2 in N), each wave 32x64 via 2x4 WMMA tiles.
template <bool BT, bool OUT_BF16>
__global__ __launch_bounds__(256)
void gemm_wmma_kernel(const unsigned short* __restrict__ Ag,
                      const unsigned short* __restrict__ Bg,
                      void* __restrict__ Cg,
                      int M, int N, int K,
                      int lda, int ldb, int ldc,
                      long long sA, long long sB, long long sC,
                      float scale)
{
    __shared__ unsigned short As[BM * LDA_S];
    __shared__ unsigned short Bs[BK * LDB_S];

    const int tid  = threadIdx.x;
    const int lane = tid & 31;
    const int wave = tid >> 5;
    const int wy   = wave & 3;    // 0..3 -> 32-row slice
    const int wx   = wave >> 2;   // 0..1 -> 64-col slice
    const int row0 = blockIdx.y * BM;
    const int col0 = blockIdx.x * BN;
    const int b    = blockIdx.z;

    Ag += (long long)b * sA;
    Bg += (long long)b * sB;

    v8f acc[2][4] = {};

    for (int k0 = 0; k0 < K; k0 += BK) {
        __syncthreads();   // previous-iteration readers done before overwrite

#if USE_TDM
        // ---- A tile (128 x 32) and NN B tile (32 x 128) via Tensor Data Mover ----
        if (wave == 0) {
            tdm_load_2d((uint32_t)(uintptr_t)(void*)As,
                        Ag + (long long)row0 * lda + k0,
                        /*tile*/ BK, BM, /*tensor*/ (uint32_t)K, (uint32_t)M,
                        (uint64_t)lda, /*pad: 64B rows, +8B*/ 3u, 1u);
            if constexpr (!BT) {
                tdm_load_2d((uint32_t)(uintptr_t)(void*)Bs,
                            Bg + (long long)k0 * ldb + col0,
                            /*tile*/ BN, BK, /*tensor*/ (uint32_t)N, (uint32_t)K,
                            (uint64_t)ldb, /*pad: 256B rows, +16B*/ 5u, 3u);
            }
        }
#else
        // ---- fallback: per-thread A staging ----
        {
            const int r = tid >> 1, seg = tid & 1;
            const uint2* src = (const uint2*)(Ag + (long long)(row0 + r) * lda + k0 + seg * 16);
            uint2* dst = (uint2*)(As + r * LDA_S + seg * 16);
#pragma unroll
            for (int i = 0; i < 4; ++i) dst[i] = src[i];
        }
        if constexpr (!BT) {
            const int r = tid >> 3, seg = tid & 7;
            const uint2* src = (const uint2*)(Bg + (long long)(k0 + r) * ldb + col0 + seg * 16);
            uint2* dst = (uint2*)(Bs + r * LDB_S + seg * 16);
#pragma unroll
            for (int i = 0; i < 4; ++i) dst[i] = src[i];
        }
#endif
        if constexpr (BT) {
            // B is N x K row-major: transpose while staging (manual path)
            const int n = tid >> 1, kseg = tid & 1;
            union { uint2 u2[4]; unsigned short u[16]; } tmp;
            const uint2* src = (const uint2*)(Bg + (long long)(col0 + n) * ldb + k0 + kseg * 16);
#pragma unroll
            for (int i = 0; i < 4; ++i) tmp.u2[i] = src[i];
#pragma unroll
            for (int e = 0; e < 16; ++e)
                Bs[(kseg * 16 + e) * LDB_S + n] = tmp.u[e];
            if (k0 + BK < K)  // speculative prefetch of next B chunk
                __builtin_prefetch(Bg + (long long)(col0 + n) * ldb + k0 + BK, 0, 0);
        }

#if USE_TDM
        if (wave == 0) __builtin_amdgcn_s_wait_tensorcnt(0);
#endif
        __syncthreads();

        v16bf af[2], bfv[4];
#pragma unroll
        for (int mi = 0; mi < 2; ++mi) af[mi] = ld_frag_a(As, wy * 32 + mi * 16, lane);
#pragma unroll
        for (int ni = 0; ni < 4; ++ni) bfv[ni] = ld_frag_b(Bs, wx * 64 + ni * 16, lane);
#pragma unroll
        for (int mi = 0; mi < 2; ++mi)
#pragma unroll
            for (int ni = 0; ni < 4; ++ni)
                acc[mi][ni] = __builtin_amdgcn_wmma_f32_16x16x32_bf16(
                    false, af[mi], false, bfv[ni], (short)0, acc[mi][ni], false, false);
    }

    // ---- epilogue: C layout row = r + 8*(lane/16), col = lane%16 ----
    const int hi = lane >> 4, lo = lane & 15;
#pragma unroll
    for (int mi = 0; mi < 2; ++mi) {
#pragma unroll
        for (int ni = 0; ni < 4; ++ni) {
            const int rr = row0 + wy * 32 + mi * 16 + 8 * hi;
            const int cc = col0 + wx * 64 + ni * 16 + lo;
            if constexpr (OUT_BF16) {
                unsigned short* C = (unsigned short*)Cg + (long long)b * sC;
#pragma unroll
                for (int r = 0; r < 8; ++r)
                    C[(long long)(rr + r) * ldc + cc] = f32_to_bf16(acc[mi][ni][r] * scale);
            } else {
                float* C = (float*)Cg + (long long)b * sC;
#pragma unroll
                for (int r = 0; r < 8; ++r)
                    C[(long long)(rr + r) * ldc + cc] = acc[mi][ni][r] * scale;
            }
        }
    }
}

// ---------------- row softmax: f32 scores -> bf16 probabilities ----------------
__global__ __launch_bounds__(256)
void softmax_kernel(const float* __restrict__ S, unsigned short* __restrict__ P) {
    __shared__ float red[256];
    const long long row = blockIdx.x;
    const float* s = S + row * SEQ;
    unsigned short* p = P + row * SEQ;
    const int t = threadIdx.x;

    float v[8];
#pragma unroll
    for (int i = 0; i < 8; ++i) v[i] = s[t * 8 + i];

    float m = v[0];
#pragma unroll
    for (int i = 1; i < 8; ++i) m = fmaxf(m, v[i]);
    red[t] = m; __syncthreads();
    for (int off = 128; off > 0; off >>= 1) {
        if (t < off) red[t] = fmaxf(red[t], red[t + off]);
        __syncthreads();
    }
    m = red[0]; __syncthreads();

    float sum = 0.f;
#pragma unroll
    for (int i = 0; i < 8; ++i) { v[i] = __expf(v[i] - m); sum += v[i]; }
    red[t] = sum; __syncthreads();
    for (int off = 128; off > 0; off >>= 1) {
        if (t < off) red[t] += red[t + off];
        __syncthreads();
    }
    const float inv = 1.0f / red[0];
#pragma unroll
    for (int i = 0; i < 8; ++i) p[t * 8 + i] = f32_to_bf16(v[i] * inv);
}

// ---------------- launcher ----------------
extern "C" void kernel_launch(void* const* d_in, const int* in_sizes, int n_in,
                              void* d_out, int out_size, void* d_ws, size_t ws_size,
                              hipStream_t stream) {
    const float* x  = (const float*)d_in[0];
    const float* Wq = (const float*)d_in[1];
    const float* Wk = (const float*)d_in[2];
    const float* Wv = (const float*)d_in[3];
    float* out = (float*)d_out;

    // workspace carve-up (bf16 halves unless noted)
    unsigned short* xb  = (unsigned short*)d_ws;             // BS x DIN
    unsigned short* WtQ = xb  + BS * DIN;                    // DIN x DH
    unsigned short* WtK = WtQ + (long long)DIN * DH;
    unsigned short* WtV = WtK + (long long)DIN * DH;
    unsigned short* Qb  = WtV + (long long)DIN * DH;         // BS x DH
    unsigned short* Kb  = Qb  + BS * DH;
    unsigned short* Vb  = Kb  + BS * DH;
    float*          Sc  = (float*)(Vb + BS * DH);            // BATCH x SEQ x SEQ f32
    unsigned short* Pb  = (unsigned short*)(Sc + (long long)BATCH * SEQ * SEQ);

    // 1) convert x, transpose+convert weights
    {
        const long long n = BS * DIN;
        f32_to_bf16_kernel<<<(unsigned)((n + 255) / 256), 256, 0, stream>>>(x, xb, n);
        dim3 tb(32, 8), tg(DIN / 32, DH / 32);
        transpose_w_kernel<<<tg, tb, 0, stream>>>(Wq, WtQ);
        transpose_w_kernel<<<tg, tb, 0, stream>>>(Wk, WtK);
        transpose_w_kernel<<<tg, tb, 0, stream>>>(Wv, WtV);
    }

    // 2) QKV projections: [BS x DIN] @ [DIN x DH] -> bf16
    {
        dim3 g(DH / BN, (unsigned)(BS / BM), 1);
        gemm_wmma_kernel<false, true><<<g, 256, 0, stream>>>(
            xb, WtQ, Qb, (int)BS, DH, DIN, DIN, DH, DH, 0, 0, 0, 1.0f);
        gemm_wmma_kernel<false, true><<<g, 256, 0, stream>>>(
            xb, WtK, Kb, (int)BS, DH, DIN, DIN, DH, DH, 0, 0, 0, 1.0f);
        gemm_wmma_kernel<false, true><<<g, 256, 0, stream>>>(
            xb, WtV, Vb, (int)BS, DH, DIN, DIN, DH, DH, 0, 0, 0, 1.0f);
    }

    // 3) scores[b] = scale * Q[b] @ K[b]^T  (NT), f32 out
    {
        dim3 g(SEQ / BN, SEQ / BM, BATCH);
        const float scale = 1.0f / sqrtf((float)SEQ);   // reference scales by sqrt(seq_len)
        gemm_wmma_kernel<true, false><<<g, 256, 0, stream>>>(
            Qb, Kb, Sc, SEQ, SEQ, DH, DH, DH, SEQ,
            (long long)SEQ * DH, (long long)SEQ * DH, (long long)SEQ * SEQ, scale);
    }

    // 4) row softmax -> bf16 P
    softmax_kernel<<<(unsigned)BS, 256, 0, stream>>>(Sc, Pb);

    // 5) out[b] = P[b] @ V[b]  (NN), f32 out straight to d_out
    {
        dim3 g(DH / BN, SEQ / BM, BATCH);
        gemm_wmma_kernel<false, false><<<g, 256, 0, stream>>>(
            Pb, Vb, out, SEQ, DH, SEQ, SEQ, DH, DH,
            (long long)SEQ * SEQ, (long long)SEQ * DH, (long long)SEQ * DH, 1.0f);
    }
}
